// Decoder_79173427135082
// MI455X (gfx1250) — matrix operand order repaired and checked by reading, not compiled, not run on hardware
//
#include <hip/hip_runtime.h>

#define LATENT       256
#define HIDDEN       8192
#define OUT_N        32768
#define NNZ_PER_ROW  32
#define BATCH        128

typedef __attribute__((ext_vector_type(16))) __bf16 v16bf;
typedef __attribute__((ext_vector_type(8)))  float  v8f;

// Convert 8 consecutive floats at p into bf16 lanes [base, base+8) of dst.
__device__ __forceinline__ void cvt8_to(v16bf& dst, int base, const float* __restrict__ p) {
    float4 x = *(const float4*)p;
    float4 y = *(const float4*)(p + 4);
    dst[base + 0] = (__bf16)x.x; dst[base + 1] = (__bf16)x.y;
    dst[base + 2] = (__bf16)x.z; dst[base + 3] = (__bf16)x.w;
    dst[base + 4] = (__bf16)y.x; dst[base + 5] = (__bf16)y.y;
    dst[base + 6] = (__bf16)y.z; dst[base + 7] = (__bf16)y.w;
}

// Kernel 1: h = sigmoid(w @ W1^T + b1), one wave per 16x16 (batch x hidden) tile.
// Block = 256 threads = 8 waves -> covers all 128 batch rows for one 16-wide
// hidden tile. Grid = HIDDEN/16 = 512 blocks.
__global__ __launch_bounds__(256) void gemm_sigmoid_kernel(
    const float* __restrict__ w,   // [BATCH, LATENT]
    const float* __restrict__ W1,  // [HIDDEN, LATENT]
    const float* __restrict__ b1,  // [HIDDEN]
    float* __restrict__ h)         // [BATCH, HIDDEN]
{
    const int lane  = threadIdx.x & 31;
    const int wave  = threadIdx.x >> 5;
    const int half  = lane >> 4;   // 0: lanes 0-15, 1: lanes 16-31
    const int l15   = lane & 15;
    const int mBase = wave * 16;           // batch tile (8 waves x 16 = 128)
    const int nBase = blockIdx.x * 16;     // hidden tile

    // Per-lane base pointers: A row = w row (M = l15), B column = W1 row (N = l15)
    const float* __restrict__ Arow = w  + (size_t)(mBase + l15) * LATENT;
    const float* __restrict__ Bcol = W1 + (size_t)(nBase + l15) * LATENT;

    v8f acc = {};

    #pragma unroll
    for (int k0 = 0; k0 < LATENT; k0 += 32) {
        v16bf a, b;
        // A 16x32 bf16 layout: lanes 0-15 hold K = {k0+0..7, k0+16..23},
        // lanes 16-31 hold K = {k0+8..15, k0+24..31} (per-lane contiguous chunks).
        cvt8_to(a, 0, Arow + k0 + half * 8);
        cvt8_to(a, 8, Arow + k0 + 16 + half * 8);
        // B 32x16 bf16 layout: same K order as A; lanes 0-15 K = {k0+0..7, k0+16..23},
        // lanes 16-31 K = {k0+8..15, k0+24..31}.
        cvt8_to(b, 0, Bcol + k0 + half * 8);
        cvt8_to(b, 8, Bcol + k0 + 16 + half * 8);

        acc = __builtin_amdgcn_wmma_f32_16x16x32_bf16(
            /*neg_a=*/false, a, /*neg_b=*/false, b,
            /*c_mod=*/(short)0, acc, /*reuse_a=*/false, /*reuse_b=*/false);
    }

    // C/D layout: reg r, lane l -> m = mBase + r + 8*(l>=16), n = nBase + (l&15)
    const float bias = b1[nBase + l15];
    #pragma unroll
    for (int r = 0; r < 8; ++r) {
        const int m = mBase + r + 8 * half;
        const float x = acc[r] + bias;
        const float s = 1.0f / (1.0f + __expf(-x));
        h[(size_t)m * HIDDEN + nBase + l15] = s;
    }
}

// Kernel 2: out[b, r] = scale[r] * sum_j ev[r,j] * h[b, cols[r,j]] + ref[r]
// One thread per output row r (lanes -> consecutive r => coalesced stores).
// cols/vals for the row cached in registers, reused over a 32-batch slice
// (gridDim.y = 4 slices for parallelism).
__global__ __launch_bounds__(256) void sparse_scale_kernel(
    const float* __restrict__ h,      // [BATCH, HIDDEN]
    const float* __restrict__ ev,     // [OUT_N * NNZ_PER_ROW]
    const int*   __restrict__ cols,   // [OUT_N * NNZ_PER_ROW]
    const float* __restrict__ scale,  // [OUT_N]
    const float* __restrict__ ref,    // [OUT_N]
    float* __restrict__ out)          // [BATCH, OUT_N]
{
    const int r     = blockIdx.x * blockDim.x + threadIdx.x;
    const int bBase = blockIdx.y * (BATCH / 4);

    int   c[NNZ_PER_ROW];
    float v[NNZ_PER_ROW];
    const int*   __restrict__ cp = cols + (size_t)r * NNZ_PER_ROW;
    const float* __restrict__ vp = ev   + (size_t)r * NNZ_PER_ROW;
    #pragma unroll
    for (int j = 0; j < NNZ_PER_ROW; j += 4) {
        int4   ci = *(const int4*)(cp + j);
        float4 vi = *(const float4*)(vp + j);
        c[j + 0] = ci.x; c[j + 1] = ci.y; c[j + 2] = ci.z; c[j + 3] = ci.w;
        v[j + 0] = vi.x; v[j + 1] = vi.y; v[j + 2] = vi.z; v[j + 3] = vi.w;
    }

    const float sc = scale[r];
    const float rf = ref[r];

    #pragma unroll 2
    for (int b = bBase; b < bBase + BATCH / 4; ++b) {
        const float* __restrict__ hb = h + (size_t)b * HIDDEN;
        float s = 0.0f;
        #pragma unroll
        for (int j = 0; j < NNZ_PER_ROW; ++j)
            s = fmaf(v[j], hb[c[j]], s);
        out[(size_t)b * OUT_N + r] = fmaf(sc, s, rf);
    }
}

extern "C" void kernel_launch(void* const* d_in, const int* in_sizes, int n_in,
                              void* d_out, int out_size, void* d_ws, size_t ws_size,
                              hipStream_t stream) {
    const float* w     = (const float*)d_in[0];   // [128, 256]
    const float* W1    = (const float*)d_in[1];   // [8192, 256]
    const float* b1    = (const float*)d_in[2];   // [8192]
    const float* ev    = (const float*)d_in[3];   // [OUT*32]
    // d_in[4] = rows: implicit (rows[e] == e / 32), unused
    const int*   cols  = (const int*)d_in[5];     // [OUT*32]
    const float* scale = (const float*)d_in[6];   // [OUT]
    const float* ref   = (const float*)d_in[7];   // [OUT]
    float*       out   = (float*)d_out;           // [128, 32768]
    float*       h     = (float*)d_ws;            // [128, 8192] fp32 = 4 MB scratch

    // Phase 1: dense bf16-WMMA GEMM + bias + sigmoid -> h (stays hot in L2)
    gemm_sigmoid_kernel<<<dim3(HIDDEN / 16), 256, 0, stream>>>(w, W1, b1, h);

    // Phase 2: sparse gather / scale / add (coalesced output stores)
    sparse_scale_kernel<<<dim3(OUT_N / 256, 4), 256, 0, stream>>>(
        h, ev, cols, scale, ref, out);
}
